// TransformerBlock_4690104287476
// MI455X (gfx1250) — matrix operand order repaired and checked by reading, not compiled, not run on hardware
//
#include <hip/hip_runtime.h>
#include <math.h>

typedef __attribute__((ext_vector_type(16))) __bf16 v16bf;
typedef __attribute__((ext_vector_type(8)))  __bf16 v8bf;
typedef __attribute__((ext_vector_type(8)))  float  v8f;
typedef __attribute__((ext_vector_type(4)))  float  vf4;
typedef __attribute__((ext_vector_type(4)))  unsigned int u32x4;
typedef __attribute__((ext_vector_type(8)))  int i32x8;
typedef __attribute__((ext_vector_type(4)))  int i32x4;

#define NROWS  4096   // B * S
#define DMODEL 1024
#define DFF    4096
#define SEQ    2048
#define NHEAD  16
#define HDIM   64
#define LDSS   48     // padded LDS row stride (bf16 elems); 96B keeps 16B alignment

#if defined(__has_builtin)
#if __has_builtin(__builtin_amdgcn_tensor_load_to_lds) && \
    __has_builtin(__builtin_amdgcn_s_wait_tensorcnt)
#define HAVE_TDM 1
#endif
#endif
#ifndef HAVE_TDM
#define HAVE_TDM 0
#endif

union BF16x16 { v16bf v; v8bf h[2]; };

static __device__ __forceinline__ v8f wmma_bf16(v16bf a, v16bf b, v8f c) {
  return __builtin_amdgcn_wmma_f32_16x16x32_bf16(false, a, false, b, (short)0, c,
                                                 false, false);
}

// ---------------- fp32 -> bf16 weight conversion ----------------
__global__ void cvt_bf16(const float* __restrict__ src, __bf16* __restrict__ dst, int n) {
  int i = blockIdx.x * blockDim.x + threadIdx.x;
  int stride = gridDim.x * blockDim.x;
  for (; i < n; i += stride) dst[i] = (__bf16)src[i];
}

// ---------------- RMSNorm: fp32 row -> bf16 row ----------------
__global__ __launch_bounds__(256) void rmsnorm_kernel(
    const float* __restrict__ x, const float* __restrict__ g,
    __bf16* __restrict__ out) {
  __shared__ float red[8];
  const int row = blockIdx.x, tid = threadIdx.x;
  const float* xr = x + (size_t)row * DMODEL;
  vf4 v = *(const vf4*)(xr + tid * 4);
  float ss = v.x * v.x + v.y * v.y + v.z * v.z + v.w * v.w;
#pragma unroll
  for (int o = 16; o > 0; o >>= 1) ss += __shfl_xor(ss, o, 32);
  if ((tid & 31) == 0) red[tid >> 5] = ss;
  __syncthreads();
  float tot = 0.f;
#pragma unroll
  for (int i = 0; i < 8; ++i) tot += red[i];
  float rinv = rsqrtf(tot * (1.0f / DMODEL) + 1e-5f);
  vf4 gg = *(const vf4*)(g + tid * 4);
  __bf16* orow = out + (size_t)row * DMODEL + tid * 4;
  orow[0] = (__bf16)(v.x * rinv * gg.x);
  orow[1] = (__bf16)(v.y * rinv * gg.y);
  orow[2] = (__bf16)(v.z * rinv * gg.z);
  orow[3] = (__bf16)(v.w * rinv * gg.w);
}

// ---------------- RoPE: fp32 Q/K -> bf16 (Q scaled by 1/sqrt(HDIM)) ----------------
__global__ __launch_bounds__(256) void rope_kernel(
    const float* __restrict__ Qf, const float* __restrict__ Kf,
    __bf16* __restrict__ Qb, __bf16* __restrict__ Kb) {
  const int row = blockIdx.x;
  const int s = row & (SEQ - 1);
  for (int pp = threadIdx.x; pp < DMODEL / 2; pp += 256) {
    int head = pp >> 5, pi = pp & 31;            // 32 pairs per head
    float inv = expf(-((float)(2 * pi) / HDIM) * logf(10000.0f));
    float ang = (float)s * inv;
    float c = cosf(ang), sn = sinf(ang);
    size_t base = (size_t)row * DMODEL + head * HDIM + 2 * pi;
    float q1 = Qf[base], q2 = Qf[base + 1];
    Qb[base]     = (__bf16)((q1 * c - q2 * sn) * 0.125f);
    Qb[base + 1] = (__bf16)((q2 * c + q1 * sn) * 0.125f);
    float k1 = Kf[base], k2 = Kf[base + 1];
    Kb[base]     = (__bf16)(k1 * c - k2 * sn);
    Kb[base + 1] = (__bf16)(k2 * c + k1 * sn);
  }
}

// ---------------- SwiGLU elementwise ----------------
__global__ void swiglu_kernel(const __bf16* __restrict__ x1,
                              const __bf16* __restrict__ x3,
                              __bf16* __restrict__ t, size_t n) {
  size_t i = (size_t)blockIdx.x * blockDim.x + threadIdx.x;
  size_t stride = (size_t)gridDim.x * blockDim.x;
  for (; i < n; i += stride) {
    float a = (float)x1[i];
    float b = (float)x3[i];
    t[i] = (__bf16)(a * (1.0f / (1.0f + expf(-a))) * b);
  }
}

// ---------------- bf16 WMMA GEMM: C[M,N] = A[M,K] @ B[K,N] (+ residual) ----------------
template <bool OUT_BF16, bool HAS_RES>
__global__ __launch_bounds__(256) void gemm_bf16_kernel(
    const __bf16* __restrict__ A, const __bf16* __restrict__ B,
    void* __restrict__ C, const float* __restrict__ R,
    int M, int N, int K) {
  __shared__ __bf16 As[128 * LDSS];
  __shared__ __bf16 Bs[128 * LDSS];   // transposed: Bs[n][k]
  const int tid = threadIdx.x;
  const int lane = tid & 31, wave = tid >> 5;
  const int wm = wave & 3, wn = wave >> 2;   // 4 waves on M, 2 on N
  const int lmod = lane & 15, lhalf = lane >> 4;
  const int bm = blockIdx.x * 128, bn = blockIdx.y * 128;

  v8f zero = {0.f, 0.f, 0.f, 0.f, 0.f, 0.f, 0.f, 0.f};
  v8f acc[2][4];
#pragma unroll
  for (int i = 0; i < 2; ++i)
#pragma unroll
    for (int j = 0; j < 4; ++j) acc[i][j] = zero;

  for (int k0 = 0; k0 < K; k0 += 32) {
#if HAVE_TDM
    // Stage A tile (128 rows x 32 bf16) via the Tensor Data Mover.
    // TDM pads the LDS destination: 64B row + 32B pad == LDSS (48 bf16) stride.
    if (wave == 0) {
      unsigned lds_off = (unsigned)(uintptr_t)(&As[0]);           // LDS byte addr
      unsigned long long ga =
          (unsigned long long)(uintptr_t)(A + (size_t)bm * K + k0);
      u32x4 g0;
      g0[0] = 1u;                                  // count=1, user descriptor
      g0[1] = lds_off;                             // lds_addr
      g0[2] = (unsigned)(ga & 0xffffffffull);      // global_addr[31:0]
      g0[3] = (unsigned)((ga >> 32) & 0x1ffffffull) | (2u << 30);  // addr hi | type=2
      i32x8 g1;
      g1[0] = (int)((1u << 16)   |                 // data_size = 2 bytes
                    (1u << 20)   |                 // pad_enable
                    (3u << 22)   |                 // pad_interval: 16 DWORDs (64B)
                    (7u << 25)); //                   pad_amount: 8 DWORDs (32B)
      g1[1] = (int)(((unsigned)K & 0xffffu) << 16);                 // tensor_dim0 lo
      g1[2] = (int)(((unsigned)K >> 16) | (((unsigned)M & 0xffffu) << 16)); // dim0 hi | dim1 lo
      g1[3] = (int)(((unsigned)M >> 16) | (32u << 16));             // dim1 hi | tile_dim0=32
      g1[4] = 128;                                                  // tile_dim1=128
      g1[5] = (int)K;                                               // tensor_dim0_stride lo32
      g1[6] = 0;                                                    // stride hi / dim1_stride lo
      g1[7] = 0;
      i32x4 gz = {0, 0, 0, 0};
#if __clang_major__ >= 23
      i32x8 gz8 = {0, 0, 0, 0, 0, 0, 0, 0};
      __builtin_amdgcn_tensor_load_to_lds(g0, g1, gz, gz, gz8, 0);
#else
      __builtin_amdgcn_tensor_load_to_lds(g0, g1, gz, gz, 0);
#endif
      __builtin_amdgcn_s_wait_tensorcnt(0);
    }
#else
    // A tile 128x32, row-major (manual fallback)
#pragma unroll
    for (int c = 0; c < 2; ++c) {
      int idx = tid * 2 + c;
      int r = idx >> 2, col = (idx & 3) << 3;
      *(v8bf*)(As + r * LDSS + col) =
          *(const v8bf*)(A + (size_t)(bm + r) * K + k0 + col);
    }
#endif
    // B tile 32x128 loaded row-major, stored transposed Bs[n][k]
#pragma unroll
    for (int c = 0; c < 2; ++c) {
      int idx = tid * 2 + c;
      int kr = idx >> 4, nc = (idx & 15) << 3;
      v8bf d = *(const v8bf*)(B + (size_t)(k0 + kr) * N + bn + nc);
#pragma unroll
      for (int j = 0; j < 8; ++j) Bs[(nc + j) * LDSS + kr] = d[j];
    }
    // prefetch next K-step tiles into cache
    if (k0 + 32 < K) {
      __builtin_prefetch(A + (size_t)(bm + (tid >> 1)) * K + k0 + 32, 0, 1);
      __builtin_prefetch(B + (size_t)(k0 + 32 + (tid & 31)) * N + bn, 0, 1);
    }
    __syncthreads();

    v16bf af[2], bfr[4];
#pragma unroll
    for (int mi = 0; mi < 2; ++mi) {
      const __bf16* p = As + (wm * 32 + mi * 16 + lmod) * LDSS + lhalf * 8;
      BF16x16 u; u.h[0] = *(const v8bf*)p; u.h[1] = *(const v8bf*)(p + 16);
      af[mi] = u.v;
    }
#pragma unroll
    for (int ni = 0; ni < 4; ++ni) {
      const __bf16* p = Bs + (wn * 64 + ni * 16 + lmod) * LDSS + lhalf * 8;
      BF16x16 u; u.h[0] = *(const v8bf*)p; u.h[1] = *(const v8bf*)(p + 16);
      bfr[ni] = u.v;
    }
#pragma unroll
    for (int mi = 0; mi < 2; ++mi)
#pragma unroll
      for (int ni = 0; ni < 4; ++ni)
        acc[mi][ni] = wmma_bf16(af[mi], bfr[ni], acc[mi][ni]);
    __syncthreads();
  }

#pragma unroll
  for (int mi = 0; mi < 2; ++mi)
#pragma unroll
    for (int ni = 0; ni < 4; ++ni)
#pragma unroll
      for (int j = 0; j < 8; ++j) {
        int r = bm + wm * 32 + mi * 16 + j + lhalf * 8;
        int cc = bn + wn * 64 + ni * 16 + lmod;
        size_t o = (size_t)r * N + cc;
        float v = acc[mi][ni][j];
        if (HAS_RES) v += R[o];
        if (OUT_BF16) ((__bf16*)C)[o] = (__bf16)v;
        else          ((float*)C)[o] = v;
      }
}

// ---------------- Flash attention (causal, per-head) ----------------
// grid (S/64, NHEAD, B), block 128 (4 waves, 16 query rows each)
__global__ __launch_bounds__(128) void flash_kernel(
    const __bf16* __restrict__ Qb, const __bf16* __restrict__ Kb,
    const __bf16* __restrict__ Vb, __bf16* __restrict__ Ctx) {
  __shared__ __bf16 Vt[HDIM * LDSS];    // Vt[d][key], transposed V tile
  __shared__ __bf16 Pl[4][16 * LDSS];   // per-wave probability tile
  const int qt = blockIdx.x, hh = blockIdx.y, b = blockIdx.z;
  const int tid = threadIdx.x, lane = tid & 31, wave = tid >> 5;
  const int lmod = lane & 15, lhalf = lane >> 4;
  const int hoff = hh * HDIM;

  const int qrow = qt * 64 + wave * 16 + lmod;
  const size_t qbase = ((size_t)b * SEQ + qrow) * DMODEL + hoff;
  v16bf aq[2];
#pragma unroll
  for (int f = 0; f < 2; ++f) {
    const __bf16* p = Qb + qbase + f * 32 + lhalf * 8;
    BF16x16 u; u.h[0] = *(const v8bf*)p; u.h[1] = *(const v8bf*)(p + 16);
    aq[f] = u.v;
  }

  v8f zero = {0.f, 0.f, 0.f, 0.f, 0.f, 0.f, 0.f, 0.f};
  v8f acc[4];
  float mrow[8], lrow[8];
#pragma unroll
  for (int n = 0; n < 4; ++n) acc[n] = zero;
#pragma unroll
  for (int j = 0; j < 8; ++j) { mrow[j] = -INFINITY; lrow[j] = 0.f; }

  const int kblocks = qt * 2 + 2;   // keys up to (qt+1)*64
  for (int kb = 0; kb < kblocks; ++kb) {
    const int k0 = kb * 32;
    // cooperative load of V tile (32 keys x 64 dims) transposed into LDS
    {
      int key = tid & 31, ds = tid >> 5;   // ds: 0..3, 16 dims each
      const __bf16* vp = Vb + ((size_t)b * SEQ + k0 + key) * DMODEL + hoff + ds * 16;
      v8bf d0 = *(const v8bf*)vp;
      v8bf d1 = *(const v8bf*)(vp + 8);
#pragma unroll
      for (int j = 0; j < 8; ++j) {
        Vt[(ds * 16 + j) * LDSS + key] = d0[j];
        Vt[(ds * 16 + 8 + j) * LDSS + key] = d1[j];
      }
    }
    // scores: two 16x16 tiles over 32 keys
    v8f sc[2];
#pragma unroll
    for (int t = 0; t < 2; ++t) {
      const int key = k0 + t * 16 + lmod;
      const __bf16* kp = Kb + ((size_t)b * SEQ + key) * DMODEL + hoff + lhalf * 8;
      BF16x16 u0, u1;
      u0.h[0] = *(const v8bf*)kp;        u0.h[1] = *(const v8bf*)(kp + 16);
      u1.h[0] = *(const v8bf*)(kp + 32); u1.h[1] = *(const v8bf*)(kp + 48);
      v8f s = zero;
      s = wmma_bf16(aq[0], u0.v, s);
      s = wmma_bf16(aq[1], u1.v, s);
#pragma unroll
      for (int j = 0; j < 8; ++j) {
        int qg = qt * 64 + wave * 16 + j + lhalf * 8;
        int kg = k0 + t * 16 + lmod;
        if (kg > qg) s[j] = -INFINITY;
      }
      sc[t] = s;
    }
    // online softmax: reduce across the 16 lanes holding one row
    float pr0[8], pr1[8];
#pragma unroll
    for (int j = 0; j < 8; ++j) {
      float rm = fmaxf(sc[0][j], sc[1][j]);
#pragma unroll
      for (int o = 1; o < 16; o <<= 1) rm = fmaxf(rm, __shfl_xor(rm, o, 16));
      float nm = fmaxf(mrow[j], rm);
      float alpha = expf(mrow[j] - nm);          // 0 on first block
      float p0 = expf(sc[0][j] - nm);
      float p1 = expf(sc[1][j] - nm);
      float psum = p0 + p1;
#pragma unroll
      for (int o = 1; o < 16; o <<= 1) psum += __shfl_xor(psum, o, 16);
      lrow[j] = lrow[j] * alpha + psum;
      mrow[j] = nm;
#pragma unroll
      for (int n = 0; n < 4; ++n) acc[n][j] *= alpha;
      pr0[j] = p0; pr1[j] = p1;
    }
    // C-layout -> A-layout transpose of P via LDS
#pragma unroll
    for (int j = 0; j < 8; ++j) {
      int rl = j + lhalf * 8;
      Pl[wave][rl * LDSS + lmod]      = (__bf16)pr0[j];
      Pl[wave][rl * LDSS + 16 + lmod] = (__bf16)pr1[j];
    }
    __syncthreads();
    BF16x16 up;
    const __bf16* pp = &Pl[wave][lmod * LDSS + lhalf * 8];
    up.h[0] = *(const v8bf*)pp; up.h[1] = *(const v8bf*)(pp + 16);
#pragma unroll
    for (int n = 0; n < 4; ++n) {
      const __bf16* vv = &Vt[(n * 16 + lmod) * LDSS + lhalf * 8];
      BF16x16 uv; uv.h[0] = *(const v8bf*)vv; uv.h[1] = *(const v8bf*)(vv + 16);
      acc[n] = wmma_bf16(up.v, uv.v, acc[n]);
    }
    __syncthreads();
  }
  // normalize and store bf16 context
#pragma unroll
  for (int n = 0; n < 4; ++n)
#pragma unroll
    for (int j = 0; j < 8; ++j) {
      int rl = j + lhalf * 8;
      size_t o = ((size_t)b * SEQ + qt * 64 + wave * 16 + rl) * DMODEL +
                 hoff + n * 16 + lmod;
      Ctx[o] = (__bf16)(acc[n][j] / lrow[j]);
    }
}

// ---------------- host-side orchestration ----------------
extern "C" void kernel_launch(void* const* d_in, const int* in_sizes, int n_in,
                              void* d_out, int out_size, void* d_ws, size_t ws_size,
                              hipStream_t stream) {
  const float* x  = (const float*)d_in[0];
  const float* Wq = (const float*)d_in[1];
  const float* Wk = (const float*)d_in[2];
  const float* Wv = (const float*)d_in[3];
  const float* Wo = (const float*)d_in[4];
  const float* g1 = (const float*)d_in[5];
  const float* g2 = (const float*)d_in[6];
  const float* W1 = (const float*)d_in[7];
  const float* W2 = (const float*)d_in[8];
  const float* W3 = (const float*)d_in[9];

  char* w = (char*)d_ws;
  const size_t MB = (size_t)1 << 20;
  __bf16* Wq_b = (__bf16*)(w + 0 * MB);
  __bf16* Wk_b = (__bf16*)(w + 2 * MB);
  __bf16* Wv_b = (__bf16*)(w + 4 * MB);
  __bf16* Wo_b = (__bf16*)(w + 6 * MB);
  __bf16* W1_b = (__bf16*)(w + 8 * MB);
  __bf16* W3_b = (__bf16*)(w + 16 * MB);
  __bf16* W2_b = (__bf16*)(w + 24 * MB);
  __bf16* H1   = (__bf16*)(w + 32 * MB);
  float*  Qf   = (float*)(w + 40 * MB);
  float*  Kf   = (float*)(w + 56 * MB);
  __bf16* Qb   = (__bf16*)(w + 72 * MB);
  __bf16* Kb   = (__bf16*)(w + 80 * MB);
  __bf16* Vb   = (__bf16*)(w + 88 * MB);
  __bf16* Ctx  = (__bf16*)(w + 96 * MB);
  float*  Out1 = (float*)(w + 104 * MB);
  __bf16* H2   = (__bf16*)(w + 120 * MB);
  __bf16* X1   = (__bf16*)(w + 128 * MB);
  __bf16* X3   = (__bf16*)(w + 160 * MB);
  __bf16* T    = (__bf16*)(w + 192 * MB);

  // weight conversion
  cvt_bf16<<<2048, 256, 0, stream>>>(Wq, Wq_b, DMODEL * DMODEL);
  cvt_bf16<<<2048, 256, 0, stream>>>(Wk, Wk_b, DMODEL * DMODEL);
  cvt_bf16<<<2048, 256, 0, stream>>>(Wv, Wv_b, DMODEL * DMODEL);
  cvt_bf16<<<2048, 256, 0, stream>>>(Wo, Wo_b, DMODEL * DMODEL);
  cvt_bf16<<<2048, 256, 0, stream>>>(W1, W1_b, DMODEL * DFF);
  cvt_bf16<<<2048, 256, 0, stream>>>(W3, W3_b, DMODEL * DFF);
  cvt_bf16<<<2048, 256, 0, stream>>>(W2, W2_b, DFF * DMODEL);

  dim3 gB(256);
  dim3 g1024(NROWS / 128, DMODEL / 128);
  dim3 g4096(NROWS / 128, DFF / 128);

  rmsnorm_kernel<<<NROWS, 256, 0, stream>>>(x, g1, H1);
  gemm_bf16_kernel<false, false><<<g1024, gB, 0, stream>>>(H1, Wq_b, Qf, nullptr, NROWS, DMODEL, DMODEL);
  gemm_bf16_kernel<false, false><<<g1024, gB, 0, stream>>>(H1, Wk_b, Kf, nullptr, NROWS, DMODEL, DMODEL);
  gemm_bf16_kernel<true,  false><<<g1024, gB, 0, stream>>>(H1, Wv_b, Vb, nullptr, NROWS, DMODEL, DMODEL);
  rope_kernel<<<NROWS, 256, 0, stream>>>(Qf, Kf, Qb, Kb);
  flash_kernel<<<dim3(SEQ / 64, NHEAD, 2), 128, 0, stream>>>(Qb, Kb, Vb, Ctx);
  gemm_bf16_kernel<false, true><<<g1024, gB, 0, stream>>>(Ctx, Wo_b, Out1, x, NROWS, DMODEL, DMODEL);
  rmsnorm_kernel<<<NROWS, 256, 0, stream>>>(Out1, g2, H2);
  gemm_bf16_kernel<true, false><<<g4096, gB, 0, stream>>>(H2, W1_b, X1, nullptr, NROWS, DFF, DMODEL);
  gemm_bf16_kernel<true, false><<<g4096, gB, 0, stream>>>(H2, W3_b, X3, nullptr, NROWS, DFF, DMODEL);
  swiglu_kernel<<<4096, 256, 0, stream>>>(X1, X3, T, (size_t)NROWS * DFF);
  gemm_bf16_kernel<false, true><<<g1024, gB, 0, stream>>>(T, W2_b, (float*)d_out, Out1, NROWS, DMODEL, DFF);
}
